// MultiHeadSelfAttention_2_44272522887188
// MI455X (gfx1250) — matrix operand order repaired and checked by reading, not compiled, not run on hardware
//
#include <hip/hip_runtime.h>

typedef __bf16 bf16_t;
typedef bf16_t v16bf __attribute__((ext_vector_type(16)));
typedef float  v8f   __attribute__((ext_vector_type(8)));

#define D_MODEL 1024
#define LSEQ    2048
#define NB      2
#define NH      16
#define DK      64
#define MTOT    (NB * LSEQ)   // 4096 rows total (B*L)

union FragBF {
    v16bf v;
    uint4 q[2];
};

__device__ __forceinline__ bf16_t f2bf(float f) { return (bf16_t)f; }

// ---------------------------------------------------------------------------
// Kernel 1: QKV projection GEMM.  C[4096,1024] = A @ W + b
//   blockIdx.z: 0 -> Q (A=cdd),  1 -> K (A=his),  2 -> V (A=his)
//   Block tile 64(M) x 128(N), 4 waves, each wave 32x64 (2x4 WMMA tiles).
//   K staged in 64-wide slabs (2 x 32-K WMMA substeps per barrier pair);
//   A bf16 row-major in LDS, W bf16 transposed [n][k].
// ---------------------------------------------------------------------------
__global__ __launch_bounds__(128)
void qkv_gemm_kernel(const float* __restrict__ cdd,
                     const float* __restrict__ his,
                     const float* __restrict__ Wq, const float* __restrict__ bq,
                     const float* __restrict__ Wk, const float* __restrict__ bk,
                     const float* __restrict__ Wv, const float* __restrict__ bv,
                     float*  __restrict__ qa_out,
                     bf16_t* __restrict__ Qw,
                     bf16_t* __restrict__ Kw,
                     bf16_t* __restrict__ Vw)
{
    const int qid = blockIdx.z;
    const float* A    = (qid == 0) ? cdd : his;
    const float* W    = (qid == 0) ? Wq  : (qid == 1) ? Wk : Wv;
    const float* bias = (qid == 0) ? bq  : (qid == 1) ? bk : bv;
    bf16_t* Obf       = (qid == 0) ? Qw  : (qid == 1) ? Kw : Vw;

    const int Mbase = blockIdx.y * 64;
    const int Nbase = blockIdx.x * 128;

    __shared__ __align__(16) bf16_t As[64 * 64];    // [m][k]   8 KB
    __shared__ __align__(16) bf16_t Ws[128 * 64];   // [n][k]  16 KB (transposed)

    const int t    = threadIdx.x;
    const int lane = t & 31;
    const int w    = t >> 5;      // wave 0..3
    const int wm   = w >> 1;      // 0..1 -> M offset wm*32
    const int wn   = w & 1;       // 0..1 -> N offset wn*64
    const int lm   = lane & 15;
    const int lk   = lane >> 4;

    v8f acc[2][4];
    const v8f vzero = {0.f, 0.f, 0.f, 0.f, 0.f, 0.f, 0.f, 0.f};
#pragma unroll
    for (int mi = 0; mi < 2; ++mi)
#pragma unroll
        for (int ni = 0; ni < 4; ++ni)
            acc[mi][ni] = vzero;

    for (int kk = 0; kk < D_MODEL; kk += 64) {
        // stage A tile: 64x64 fp32 -> bf16 (1024 float4, 8 per thread)
#pragma unroll
        for (int i = 0; i < 8; ++i) {
            int e4  = t + i * 128;
            int row = e4 >> 4;            // 16 float4 per 64-wide row
            int c4  = (e4 & 15) * 4;
            float4 f = *reinterpret_cast<const float4*>(
                &A[(size_t)(Mbase + row) * D_MODEL + kk + c4]);
            As[row * 64 + c4 + 0] = f2bf(f.x);
            As[row * 64 + c4 + 1] = f2bf(f.y);
            As[row * 64 + c4 + 2] = f2bf(f.z);
            As[row * 64 + c4 + 3] = f2bf(f.w);
        }
        // stage W tile 64x128, transposed into [n][k] (2048 float4, 16/thread)
#pragma unroll
        for (int i = 0; i < 16; ++i) {
            int e4  = t + i * 128;
            int row = e4 >> 5;            // 32 float4 per 128-wide row
            int c4  = (e4 & 31) * 4;
            float4 f = *reinterpret_cast<const float4*>(
                &W[(size_t)(kk + row) * D_MODEL + Nbase + c4]);
            Ws[(c4 + 0) * 64 + row] = f2bf(f.x);
            Ws[(c4 + 1) * 64 + row] = f2bf(f.y);
            Ws[(c4 + 2) * 64 + row] = f2bf(f.z);
            Ws[(c4 + 3) * 64 + row] = f2bf(f.w);
        }
        // prefetch next slab while this one is consumed
        if (kk + 64 < D_MODEL) {
            int r = t & 63;
            __builtin_prefetch(&A[(size_t)(Mbase + r) * D_MODEL + kk + 64], 0, 1);
            __builtin_prefetch(&W[(size_t)(kk + 64 + r) * D_MODEL + Nbase + (t >> 6) * 64], 0, 1);
        }
        __syncthreads();

#pragma unroll
        for (int ks = 0; ks < 2; ++ks) {
            FragBF afr[2], bfr[4];
#pragma unroll
            for (int mi = 0; mi < 2; ++mi) {
                int base = (wm * 32 + mi * 16 + lm) * 64 + ks * 32;
                afr[mi].q[0] = *reinterpret_cast<const uint4*>(&As[base + lk * 8]);
                afr[mi].q[1] = *reinterpret_cast<const uint4*>(&As[base + 16 + lk * 8]);
            }
#pragma unroll
            for (int ni = 0; ni < 4; ++ni) {
                int base = (wn * 64 + ni * 16 + lm) * 64 + ks * 32 + lk * 8;
                bfr[ni].q[0] = *reinterpret_cast<const uint4*>(&Ws[base]);
                bfr[ni].q[1] = *reinterpret_cast<const uint4*>(&Ws[base + 16]);
            }
#pragma unroll
            for (int mi = 0; mi < 2; ++mi)
#pragma unroll
                for (int ni = 0; ni < 4; ++ni)
                    acc[mi][ni] = __builtin_amdgcn_wmma_f32_16x16x32_bf16(
                        false, afr[mi].v, false, bfr[ni].v,
                        (short)0, acc[mi][ni], false, false);
        }
        __syncthreads();
    }

    // epilogue: bias add + stores; qid branch hoisted (block-uniform)
    if (qid == 0) {
#pragma unroll
        for (int mi = 0; mi < 2; ++mi)
#pragma unroll
            for (int ni = 0; ni < 4; ++ni) {
                int gcol = Nbase + wn * 64 + ni * 16 + lm;
                float bv_ = bias[gcol];
#pragma unroll
                for (int j = 0; j < 8; ++j) {
                    int grow = Mbase + wm * 32 + mi * 16 + lk * 8 + j;
                    float val = acc[mi][ni][j] + bv_;
                    size_t idx = (size_t)grow * D_MODEL + gcol;
                    Obf[idx]    = f2bf(val);
                    qa_out[idx] = val;
                }
            }
    } else {
#pragma unroll
        for (int mi = 0; mi < 2; ++mi)
#pragma unroll
            for (int ni = 0; ni < 4; ++ni) {
                int gcol = Nbase + wn * 64 + ni * 16 + lm;
                float bv_ = bias[gcol];
#pragma unroll
                for (int j = 0; j < 8; ++j) {
                    int grow = Mbase + wm * 32 + mi * 16 + lk * 8 + j;
                    Obf[(size_t)grow * D_MODEL + gcol] = f2bf(acc[mi][ni][j] + bv_);
                }
            }
    }
}

// ---------------------------------------------------------------------------
// Kernel 2: per-head attention with unnormalized exp-softmax (streaming).
//   Block = (b, h, 64 q rows), 4 waves x 16 q rows each.
//   Per 32-key chunk: S = Q K^T (4 wmma), exp, rowsum (shfl_xor),
//   P bf16 -> LDS, context += P @ V (4 wmma, V staged transposed in LDS).
//   No rescaling needed: reference softmax has no max subtraction.
// ---------------------------------------------------------------------------
__global__ __launch_bounds__(128)
void attention_kernel(const bf16_t* __restrict__ Qw,
                      const bf16_t* __restrict__ Kw,
                      const bf16_t* __restrict__ Vw,
                      float* __restrict__ ctx)
{
    const int b     = blockIdx.y >> 4;
    const int h     = blockIdx.y & 15;
    const int qbase = blockIdx.x * 64;

    __shared__ __align__(16) bf16_t Vl[DK * 32];       // transposed [d][kpos]
    __shared__ __align__(16) bf16_t Pl[4 * 16 * 32];   // per-wave [m][kpos]

    const int t    = threadIdx.x;
    const int lane = t & 31;
    const int w    = t >> 5;
    const int lm   = lane & 15;
    const int lk   = lane >> 4;

    // Q fragments for this wave's 16 rows (A-layout, loaded once)
    const size_t rowQ = (size_t)(b * LSEQ + qbase + w * 16 + lm) * D_MODEL + h * DK;
    FragBF qf[2];
#pragma unroll
    for (int dh = 0; dh < 2; ++dh) {
        qf[dh].q[0] = *reinterpret_cast<const uint4*>(&Qw[rowQ + dh * 32 + lk * 8]);
        qf[dh].q[1] = *reinterpret_cast<const uint4*>(&Qw[rowQ + dh * 32 + 16 + lk * 8]);
    }

    const v8f vzero = {0.f, 0.f, 0.f, 0.f, 0.f, 0.f, 0.f, 0.f};
    v8f cacc[4];
#pragma unroll
    for (int nt = 0; nt < 4; ++nt) cacc[nt] = vzero;
    float rs[8];
#pragma unroll
    for (int j = 0; j < 8; ++j) rs[j] = 0.f;

    for (int kc = 0; kc < LSEQ; kc += 32) {
        __syncthreads();
        // stage V chunk 32x64 bf16 -> LDS transposed [d][kpos]
#pragma unroll
        for (int i = 0; i < 2; ++i) {
            int e8  = t + i * 128;         // 256 x (8 bf16)
            int row = e8 >> 3;             // 8 uint4 per 64-wide row
            int c8  = (e8 & 7) * 8;
            uint4 u = *reinterpret_cast<const uint4*>(
                &Vw[(size_t)(b * LSEQ + kc + row) * D_MODEL + h * DK + c8]);
            const bf16_t* pv = reinterpret_cast<const bf16_t*>(&u);
#pragma unroll
            for (int k = 0; k < 8; ++k) Vl[(c8 + k) * 32 + row] = pv[k];
        }
        // prefetch next K/V chunks (cacheline granularity per lane)
        if (kc + 32 < LSEQ) {
            int r = t & 31;
            int cq = (t >> 5) * 16;
            __builtin_prefetch(&Kw[(size_t)(b * LSEQ + kc + 32 + r) * D_MODEL + h * DK + cq], 0, 1);
            __builtin_prefetch(&Vw[(size_t)(b * LSEQ + kc + 32 + r) * D_MODEL + h * DK + cq], 0, 1);
        }
        __syncthreads();

        // S tile 16x32 = Q(16x64) @ K^T : 2 n-tiles x 2 d-halves
        v8f s0 = vzero, s1 = vzero;
#pragma unroll
        for (int dh = 0; dh < 2; ++dh) {
#pragma unroll
            for (int nt = 0; nt < 2; ++nt) {
                int kpos = kc + nt * 16 + lm;
                size_t kb = (size_t)(b * LSEQ + kpos) * D_MODEL + h * DK + dh * 32 + lk * 8;
                FragBF kf;
                kf.q[0] = *reinterpret_cast<const uint4*>(&Kw[kb]);
                kf.q[1] = *reinterpret_cast<const uint4*>(&Kw[kb + 16]);
                if (nt == 0)
                    s0 = __builtin_amdgcn_wmma_f32_16x16x32_bf16(
                        false, qf[dh].v, false, kf.v, (short)0, s0, false, false);
                else
                    s1 = __builtin_amdgcn_wmma_f32_16x16x32_bf16(
                        false, qf[dh].v, false, kf.v, (short)0, s1, false, false);
            }
        }

        // exp(s/8), rowsum via xor-reduction within 16-lane halves, P -> LDS
#pragma unroll
        for (int j = 0; j < 8; ++j) {
            float e0 = __expf(s0[j] * 0.125f);
            float e1 = __expf(s1[j] * 0.125f);
            float red = e0 + e1;
            red += __shfl_xor(red, 1, 32);
            red += __shfl_xor(red, 2, 32);
            red += __shfl_xor(red, 4, 32);
            red += __shfl_xor(red, 8, 32);
            rs[j] += red;
            int ml = lk * 8 + j;
            Pl[(w * 16 + ml) * 32 + lm]      = f2bf(e0);
            Pl[(w * 16 + ml) * 32 + 16 + lm] = f2bf(e1);
        }
        __syncthreads();

        // context += P(16x32) @ Vchunk(32x64)
        FragBF pf;
        int pb = (w * 16 + lm) * 32;
        pf.q[0] = *reinterpret_cast<const uint4*>(&Pl[pb + lk * 8]);
        pf.q[1] = *reinterpret_cast<const uint4*>(&Pl[pb + 16 + lk * 8]);
#pragma unroll
        for (int nt = 0; nt < 4; ++nt) {
            int vb = (nt * 16 + lm) * 32 + lk * 8;
            FragBF vf;
            vf.q[0] = *reinterpret_cast<const uint4*>(&Vl[vb]);
            vf.q[1] = *reinterpret_cast<const uint4*>(&Vl[vb + 16]);
            cacc[nt] = __builtin_amdgcn_wmma_f32_16x16x32_bf16(
                false, pf.v, false, vf.v, (short)0, cacc[nt], false, false);
        }
    }

    // normalize and store context fp32
#pragma unroll
    for (int nt = 0; nt < 4; ++nt) {
#pragma unroll
        for (int j = 0; j < 8; ++j) {
            int ml   = lk * 8 + j;
            int qrow = qbase + w * 16 + ml;
            int col  = h * DK + nt * 16 + lm;
            float denom = rs[j] + 1e-8f;
            ctx[(size_t)(b * LSEQ + qrow) * D_MODEL + col] = cacc[nt][j] / denom;
        }
    }
}

// ---------------------------------------------------------------------------
extern "C" void kernel_launch(void* const* d_in, const int* in_sizes, int n_in,
                              void* d_out, int out_size, void* d_ws, size_t ws_size,
                              hipStream_t stream)
{
    const float* cdd = (const float*)d_in[0];
    const float* his = (const float*)d_in[1];
    const float* Wq  = (const float*)d_in[2];
    const float* bq  = (const float*)d_in[3];
    const float* Wk  = (const float*)d_in[4];
    const float* bk  = (const float*)d_in[5];
    const float* Wv  = (const float*)d_in[6];
    const float* bv  = (const float*)d_in[7];

    float* ctx = (float*)d_out;                       // context [2,2048,1024]
    float* qa  = ctx + (size_t)MTOT * D_MODEL;        // q_a     [2,2048,1024]

    bf16_t* Qw = (bf16_t*)d_ws;                       // 8 MB each, 24 MB total
    bf16_t* Kw = Qw + (size_t)MTOT * D_MODEL;
    bf16_t* Vw = Kw + (size_t)MTOT * D_MODEL;

    // QKV projections: N tiles (1024/128=8) x M tiles (4096/64=64) x {Q,K,V}
    qkv_gemm_kernel<<<dim3(8, 64, 3), 128, 0, stream>>>(
        cdd, his, Wq, bq, Wk, bk, Wv, bv, qa, Qw, Kw, Vw);

    // attention: 32 q-tiles x (B*H = 32)
    attention_kernel<<<dim3(LSEQ / 64, NB * NH), 128, 0, stream>>>(Qw, Kw, Vw, ctx);
}
